// MaskBiGRU_51333449122321
// MI455X (gfx1250) — matrix operand order repaired and checked by reading, not compiled, not run on hardware
//
#include <hip/hip_runtime.h>
#include <hip/hip_bf16.h>

typedef __attribute__((ext_vector_type(16))) __bf16 v16bf;
typedef __attribute__((ext_vector_type(8)))  float  v8f;
typedef __attribute__((ext_vector_type(4)))  int    v4i;

#define T_LEN 512
#define BATCH 64
#define I_DIM 512
#define H_DIM 256
#define G_DIM 768   // 3*H
#define HPAD  264   // padded bf16 h row stride (halfs): shifts banks by 4/row

#if __has_builtin(__builtin_amdgcn_global_load_async_to_lds_b128) && \
    __has_builtin(__builtin_amdgcn_s_wait_asynccnt)
#define USE_ASYNC 1
typedef __attribute__((address_space(1))) v4i* g_v4i_p;   // global int4*
typedef __attribute__((address_space(3))) v4i* l_v4i_p;   // LDS int4*
#else
#define USE_ASYNC 0
#endif

union FragBF {
  v16bf v;
  uint4 q[2];
};

__device__ __forceinline__ unsigned short f2bf(float f) {
  unsigned int u = __float_as_uint(f);
  unsigned int r = u + 0x7FFFu + ((u >> 16) & 1u);   // round-to-nearest-even
  return (unsigned short)(r >> 16);
}
__device__ __forceinline__ float fsigmoid(float x) {
  return 1.0f / (1.0f + __expf(-x));
}
__device__ __forceinline__ float ftanh_fast(float x) {
  return 1.0f - 2.0f / (__expf(2.0f * x) + 1.0f);
}

// ---------------- f32 -> bf16 convert (grid-stride) ----------------
__global__ void cvt_bf16_kernel(const float* __restrict__ src,
                                unsigned short* __restrict__ dst, int n) {
  int i = blockIdx.x * blockDim.x + threadIdx.x;
  int stride = gridDim.x * blockDim.x;
  for (; i < n; i += stride) dst[i] = f2bf(src[i]);
}

// ---------------- Phase 1: gi[dir] = X @ W_ih[dir]^T + b_ih[dir] ----------------
__global__ void __launch_bounds__(256)
gi_gemm_kernel(const unsigned short* __restrict__ Xb,
               const unsigned short* __restrict__ Wih,   // [2][G_DIM*I_DIM]
               const float* __restrict__ bih1,
               const float* __restrict__ bih2,
               float* __restrict__ gi)                    // [2][T*B*G_DIM]
{
  const int dir  = blockIdx.z;
  const int m0   = blockIdx.x * 16;
  const int wave = threadIdx.x >> 5;
  const int lane = threadIdx.x & 31;
  const int nl   = lane & 15;
  const int hi   = lane >> 4;
  const int n0   = blockIdx.y * 128 + wave * 16;

  const unsigned short* W   = Wih + (size_t)dir * G_DIM * I_DIM;
  const float*          bih = dir ? bih2 : bih1;
  float*                giD = gi  + (size_t)dir * T_LEN * BATCH * G_DIM;

  const unsigned short* arow = Xb + (size_t)(m0 + nl) * I_DIM;
  const unsigned short* brow = W  + (size_t)(n0 + nl) * I_DIM;

  v8f acc;
  #pragma unroll
  for (int e = 0; e < 8; ++e) acc[e] = 0.0f;

  for (int k0 = 0; k0 < I_DIM; k0 += 32) {
    FragBF a, b;
    const int ka = k0 + hi * 16;
    a.q[0] = *(const uint4*)(arow + ka);
    a.q[1] = *(const uint4*)(arow + ka + 8);
    const int kb = k0 + hi * 16;
    b.q[0] = *(const uint4*)(brow + kb);
    b.q[1] = *(const uint4*)(brow + kb + 8);
    acc = __builtin_amdgcn_wmma_f32_16x16x32_bf16(false, a.v, false, b.v,
                                                  (short)0, acc, false, false);
  }

  const float bias = bih[n0 + nl];
  #pragma unroll
  for (int r = 0; r < 8; ++r) {
    const int m = m0 + r + hi * 8;
    giD[(size_t)m * G_DIM + n0 + nl] = acc[r] + bias;
  }
}

// ---------------- Phase 2: persistent masked GRU scan ----------------
// grid = (4 batch tiles, 2 directions), block = 256 threads (8 waves).
// Per step: async-stage gi tile into LDS (ASYNCcnt, overlaps WMMA phase),
// gh = h @ Whh^T with W tiles 0-2 register-resident / 3-5 streamed from L2,
// then fused gate math.
__global__ void __launch_bounds__(256, 1)
gru_scan_kernel(const float* __restrict__ gi,            // [2][T*B*G_DIM]
                const unsigned short* __restrict__ Whh,  // [2][G_DIM*H_DIM] bf16
                const float* __restrict__ bhh1,
                const float* __restrict__ bhh2,
                const float* __restrict__ mask,          // [T*B]
                float* __restrict__ out)                 // [T*B*512] then [2*B*256]
{
  __shared__ unsigned short h_bf[16 * HPAD];   // bf16 h for WMMA A fragments
  __shared__ float          h_f [16 * H_DIM];  // f32 master copy of h
  __shared__ float          gh  [16 * G_DIM];  // h @ Whh^T + b_hh
#if USE_ASYNC
  __shared__ float          gi_s[16 * G_DIM];  // async-staged gi tile (48 KB)
#endif

  const int dir   = blockIdx.y;
  const int brow0 = blockIdx.x * 16;
  const int tid   = threadIdx.x;
  const int wave  = tid >> 5;
  const int lane  = tid & 31;
  const int nl    = lane & 15;
  const int hi    = lane >> 4;

  const float*          giD = gi  + (size_t)dir * T_LEN * BATCH * G_DIM;
  const unsigned short* W   = Whh + (size_t)dir * G_DIM * H_DIM;
  const float*          bhh = dir ? bhh2 : bhh1;

  for (int i = tid; i < 16 * H_DIM; i += 256) h_f[i] = 0.0f;
  for (int i = tid; i < 16 * HPAD;  i += 256) h_bf[i] = 0;
  __syncthreads();

  const unsigned short* ha = &h_bf[nl * HPAD];  // this lane's A row (M = lane%16)

  // ---- register-resident W_hh fragments for this wave's tiles 0..2 ----
  FragBF wreg[3][8];
  #pragma unroll
  for (int tt = 0; tt < 3; ++tt) {
    const int n0 = wave * 96 + tt * 16;
    const unsigned short* brow = W + (size_t)(n0 + nl) * H_DIM;
    #pragma unroll
    for (int k = 0; k < 8; ++k) {
      const int kb = k * 32 + hi * 16;
      wreg[tt][k].q[0] = *(const uint4*)(brow + kb);
      wreg[tt][k].q[1] = *(const uint4*)(brow + kb + 8);
    }
  }

  for (int s = 0; s < T_LEN; ++s) {
    const int t = dir ? (T_LEN - 1 - s) : s;
    const float* giT = giD + (size_t)(t * BATCH + brow0) * G_DIM; // 12288 contiguous f32

#if USE_ASYNC
    // ---- stage gi tile into LDS on ASYNCcnt; overlaps the WMMA phase ----
    {
      char* gsrc = (char*)giT;
      #pragma unroll
      for (int c = 0; c < 12; ++c) {
        const int off = c * 4096 + tid * 16;   // 12 chunks x 4 KB, 16 B/lane
        __builtin_amdgcn_global_load_async_to_lds_b128(
            (g_v4i_p)(v4i*)(gsrc + off),
            (l_v4i_p)(v4i*)((char*)gi_s + off),
            0, 0);
      }
    }
#endif

    // ---- gh = h @ Whh^T : 8 waves x 6 tiles of 16x16, K = 256 ----
    v8f acc[6];
    #pragma unroll
    for (int tt = 0; tt < 6; ++tt)
      #pragma unroll
      for (int e = 0; e < 8; ++e) acc[tt][e] = 0.0f;

    for (int k0 = 0; k0 < H_DIM; k0 += 32) {
      const int k = k0 >> 5;
      FragBF a;
      const int ka = k0 + hi * 16;
      a.q[0] = *(const uint4*)(ha + ka);       // ds_load_b128
      a.q[1] = *(const uint4*)(ha + ka + 8);
      // tiles 0..2: register-resident B
      #pragma unroll
      for (int tt = 0; tt < 3; ++tt)
        acc[tt] = __builtin_amdgcn_wmma_f32_16x16x32_bf16(false, a.v, false,
                                                          wreg[tt][k].v,
                                                          (short)0, acc[tt],
                                                          false, false);
      // tiles 3..5: streamed from L2 (LOADcnt only; no ASYNCcnt interference)
      const int kb = k0 + hi * 16;
      #pragma unroll
      for (int tt = 3; tt < 6; ++tt) {
        const int n0 = wave * 96 + tt * 16;
        FragBF b;
        const unsigned short* brow = W + (size_t)(n0 + nl) * H_DIM + kb;
        b.q[0] = *(const uint4*)(brow);
        b.q[1] = *(const uint4*)(brow + 8);
        acc[tt] = __builtin_amdgcn_wmma_f32_16x16x32_bf16(false, a.v, false, b.v,
                                                          (short)0, acc[tt],
                                                          false, false);
      }
    }
    #pragma unroll
    for (int tt = 0; tt < 6; ++tt) {
      const int n0 = wave * 96 + tt * 16;
      const float bias = bhh[n0 + nl];
      #pragma unroll
      for (int r = 0; r < 8; ++r) {
        const int m = r + hi * 8;
        gh[m * G_DIM + n0 + nl] = acc[tt][r] + bias;
      }
    }
#if USE_ASYNC
    __builtin_amdgcn_s_wait_asynccnt(0);       // gi tile fully in LDS
#endif
    __syncthreads();

    // ---- fused gates: thread tid owns hidden column j = tid for all 16 rows ----
    const int j = tid;
    #pragma unroll 4
    for (int m = 0; m < 16; ++m) {
      const float mk = mask[t * BATCH + brow0 + m];
      const float hp = h_f[m * H_DIM + j];
#if USE_ASYNC
      const float gir = gi_s[m * G_DIM + j];
      const float giz = gi_s[m * G_DIM + 256 + j];
      const float gin = gi_s[m * G_DIM + 512 + j];
#else
      const float gir = giT[(size_t)m * G_DIM + j];
      const float giz = giT[(size_t)m * G_DIM + 256 + j];
      const float gin = giT[(size_t)m * G_DIM + 512 + j];
#endif
      const float gr = gir + gh[m * G_DIM + j];
      const float gz = giz + gh[m * G_DIM + 256 + j];
      const float hn = gh[m * G_DIM + 512 + j];
      const float r  = fsigmoid(gr);
      const float z  = fsigmoid(gz);
      const float n  = ftanh_fast(gin + r * hn);
      const float ht = (1.0f - z) * n + z * hp;
      const float hnew = mk * ht + (1.0f - mk) * hp;
      out[(size_t)(t * BATCH + brow0 + m) * 512 + dir * 256 + j] = ht * mk;
      h_f [m * H_DIM + j] = hnew;
      h_bf[m * HPAD  + j] = f2bf(hnew);
    }
    __syncthreads();
  }

  // final hidden state: hidden[dir][b][j]
  for (int m = 0; m < 16; ++m) {
    out[(size_t)T_LEN * BATCH * 512 +
        ((size_t)dir * BATCH + brow0 + m) * 256 + tid] = h_f[m * H_DIM + tid];
  }
}

extern "C" void kernel_launch(void* const* d_in, const int* in_sizes, int n_in,
                              void* d_out, int out_size, void* d_ws, size_t ws_size,
                              hipStream_t stream) {
  const float* seq_x  = (const float*)d_in[0];   // (T,B,I)
  const float* seq_m  = (const float*)d_in[1];   // (T,B)
  const float* Wih1   = (const float*)d_in[2];   // (768,512)
  const float* Whh1   = (const float*)d_in[3];   // (768,256)
  const float* bih1   = (const float*)d_in[4];
  const float* bhh1   = (const float*)d_in[5];
  const float* Wih2   = (const float*)d_in[6];
  const float* Whh2   = (const float*)d_in[7];
  const float* bih2   = (const float*)d_in[8];
  const float* bhh2   = (const float*)d_in[9];
  float* out = (float*)d_out;

  // workspace layout (256B aligned)
  char* ws = (char*)d_ws;
  unsigned short* Xb    = (unsigned short*)(ws);                               // 32 MB
  unsigned short* Wih_b = (unsigned short*)(ws + 33554432);                    // 1.5 MB
  unsigned short* Whh_b = (unsigned short*)(ws + 33554432 + 1572864);          // 0.75 MB
  float*          gi    = (float*)(ws + 33554432 + 1572864 + 786432);          // 192 MB

  const int nX   = T_LEN * BATCH * I_DIM;   // 16,777,216
  const int nWih = G_DIM * I_DIM;           // 393,216
  const int nWhh = G_DIM * H_DIM;           // 196,608

  cvt_bf16_kernel<<<8192, 256, 0, stream>>>(seq_x, Xb, nX);
  cvt_bf16_kernel<<<1024, 256, 0, stream>>>(Wih1, Wih_b,        nWih);
  cvt_bf16_kernel<<<1024, 256, 0, stream>>>(Wih2, Wih_b + nWih, nWih);
  cvt_bf16_kernel<<< 512, 256, 0, stream>>>(Whh1, Whh_b,        nWhh);
  cvt_bf16_kernel<<< 512, 256, 0, stream>>>(Whh2, Whh_b + nWhh, nWhh);

  // gi GEMM: M = T*B = 32768 (2048 tiles), N = 768 (6 groups x 8 waves), 2 dirs
  gi_gemm_kernel<<<dim3(2048, 6, 2), 256, 0, stream>>>(Xb, Wih_b, bih1, bih2, gi);

  // persistent scan: 4 batch tiles x 2 directions
  gru_scan_kernel<<<dim3(4, 2), 256, 0, stream>>>(gi, Whh_b, bhh1, bhh2, seq_m, out);
}
